// VTN_14224931685019
// MI455X (gfx1250) — matrix operand order repaired and checked by reading, not compiled, not run on hardware
//
#include <hip/hip_runtime.h>
#include <math.h>

// ---------------- problem constants ----------------
#define Bq   128
#define Fq   287
#define Eq   512
#define Sq   288          // F + 1
#define Hq   8
#define Dhq  64
#define Lq   3
#define Iq   3072
#define Mq   (Bq*Sq)      // 36864 rows
#define W1Sq 18
#define CHq  4608         // FFN row chunk = Mq/8

// ---------------- types ----------------
typedef __bf16 bf16_t;
typedef __attribute__((ext_vector_type(8)))  __bf16 v8bf;
typedef __attribute__((ext_vector_type(16))) __bf16 v16bf;
typedef __attribute__((ext_vector_type(8)))  float  v8f;
typedef unsigned v4u __attribute__((ext_vector_type(4)));
typedef int      v8i __attribute__((ext_vector_type(8)));
typedef int      v4i __attribute__((ext_vector_type(4)));

__device__ __forceinline__ unsigned bf_round_bits(float f) {
    unsigned u = __builtin_bit_cast(unsigned, f);
    return u + 0x7FFFu + ((u >> 16) & 1u);       // RNE; bf16 in bits [31:16]
}

// two f32 -> packed bf16 pair (low = first arg)
__device__ __forceinline__ unsigned pack_bf16x2(float lo, float hi) {
#if __has_builtin(__builtin_amdgcn_cvt_pk_bf16_f32)
    typedef __attribute__((ext_vector_type(2))) __bf16 v2bf;
    v2bf p = __builtin_amdgcn_cvt_pk_bf16_f32(lo, hi);
    return __builtin_bit_cast(unsigned, p);
#else
    return (bf_round_bits(lo) >> 16) | (bf_round_bits(hi) & 0xFFFF0000u);
#endif
}

__device__ __forceinline__ float gelu_exact(float x) {
    return 0.5f * x * (1.0f + erff(x * 0.70710678118654752f));
}

// =====================================================================
// Weight pre-pass: Wt[N][K] bf16 <- transpose+convert of W[K][N] f32.
// One block per 32(k) x 64(n) tile; runs once per launch per weight.
// =====================================================================
__global__ void __launch_bounds__(256)
wt_convert_kernel(const float* __restrict__ W, bf16_t* __restrict__ Wt,
                  int K, int N)
{
    __shared__ float tile[32][65];
    int kt = blockIdx.x * 32;
    int nt = blockIdx.y * 64;
    #pragma unroll
    for (int it = 0; it < 8; it++) {
        int slot = threadIdx.x + it * 256;       // 0..2047
        int kk = slot >> 6, nn = slot & 63;
        tile[kk][nn] = W[(size_t)(kt + kk) * N + nt + nn];
    }
    __syncthreads();
    #pragma unroll
    for (int it = 0; it < 4; it++) {
        int slot = threadIdx.x + it * 256;       // 0..1023 bf16 pairs
        int nn = slot >> 4;                      // 64 n-rows
        int kk = (slot & 15) * 2;                // 32 k / 2
        unsigned pk = pack_bf16x2(tile[kk][nn], tile[kk + 1][nn]);
        *(unsigned*)&Wt[(size_t)(nt + nn) * K + kt + kk] = pk;
    }
}

// =====================================================================
// Tiled GEMM: C[M,N] = act(A[M,K] @ W[K,N] + bias)
//  - A: f32, rows strided by lda, converted to bf16 (v_cvt_pk_bf16_f32)
//  - Wt: pre-converted bf16, layout [N][K]
//  - B tiles DMA'd by the Tensor Data Mover, double-buffered in LDS:
//    issue tile i+1, s_wait_tensorcnt 1, compute tile i under the DMA.
// Block tile 128x64, BK=32, 8 waves, each wave = 2x2 WMMA tiles.
// =====================================================================
#define BM 128
#define BN 64
#define BK 32
#define LDP 8   // LDS row pad -> 40 elems = 80B rows (16B aligned frags)

__global__ void __launch_bounds__(256)
gemm_bf16_wmma(const float* __restrict__ A, int lda,
               const bf16_t* __restrict__ Wt, const float* __restrict__ bias,
               float* __restrict__ C, int ldc,
               int Mrows, int N, int K, int act)
{
    __shared__ __align__(64) bf16_t sA[BM][BK + LDP];
    __shared__ __align__(64) bf16_t sB[2][BN][BK + LDP];   // sB[buf][n][k]

    const int bm   = blockIdx.x * BM;
    const int bn   = blockIdx.y * BN;
    const int tid  = threadIdx.x;
    const int wave = tid >> 5;
    const int lane = tid & 31;
    const int wm   = (wave >> 1) * 32;    // 4 waves along M
    const int wn   = (wave & 1) * 32;     // 2 waves along N
    const int mrow = lane & 15;
    const int kbA  = (lane >> 4) * 8;     // A: half-wave K runs {0-7,16-23}/{8-15,24-31}
    const int kbB  = (lane >> 4) * 8;     // B: half-wave K runs {0-7,16-23}/{8-15,24-31}

#if __has_builtin(__builtin_amdgcn_tensor_load_to_lds)
    // D# per CDNA5 ISA ch.8: 2D tile, data_size=2B, 64 rows of 64B
    // (tile_dim0=32 elems), row stride = K elems; TDM pad engine inserts
    // 4 DWORDs after every 16 DWORDs -> 80B LDS row pitch of sB[.][64][40].
    auto tdm_issue = [&](int kk0, int buf) {
        unsigned lds_base = (unsigned)(size_t)(void*)&sB[buf][0][0];
        unsigned long long ga =
            (unsigned long long)(size_t)(const void*)
                (Wt + (size_t)bn * K + kk0);
        v4u g0;
        g0[0] = 1u;                                    // count=1 valid
        g0[1] = lds_base;                              // lds_addr
        g0[2] = (unsigned)ga;                          // global_addr lo
        g0[3] = ((unsigned)(ga >> 32) & 0x01FFFFFFu)   // global_addr hi
                | (2u << 30);                          // type = image
        v8i g1;
        g1[0] = (1 << 16)      // data_size = 2 bytes
              | (1 << 20)      // pad_enable
              | (3 << 22)      // pad_interval: 16 DWORDs
              | (3 << 25);     // pad_amount:   4 DWORDs
        g1[1] = (int)(((unsigned)K & 0xFFFFu) << 16);  // tensor_dim0 lo16
        g1[2] = (int)((unsigned)K >> 16);              // tensor_dim0 hi16
        g1[3] = (16) | (32 << 16);   // tensor_dim1=1<<20 (hi16), tile_dim0=32
        g1[4] = 64;                  // tile_dim1 = 64 rows, tile_dim2 = 0
        g1[5] = K;                   // tensor_dim0_stride (lo32, elems)
        g1[6] = 0;
        g1[7] = 0;
        v4i gz; gz[0] = 0; gz[1] = 0; gz[2] = 0; gz[3] = 0;
#if __clang_major__ >= 23
        v8i gz8; gz8[0]=0; gz8[1]=0; gz8[2]=0; gz8[3]=0;
        gz8[4]=0; gz8[5]=0; gz8[6]=0; gz8[7]=0;
        __builtin_amdgcn_tensor_load_to_lds(g0, g1, gz, gz, gz8, 0);
#else
        __builtin_amdgcn_tensor_load_to_lds(g0, g1, gz, gz, 0);
#endif
    };
    if (wave == 0) tdm_issue(0, 0);      // prologue: first B tile in flight
#endif

    v8f acc[2][2];
    #pragma unroll
    for (int a = 0; a < 2; a++)
        #pragma unroll
        for (int b = 0; b < 2; b++)
            #pragma unroll
            for (int e = 0; e < 8; e++) acc[a][b][e] = 0.0f;

    int buf = 0;
    for (int k0 = 0; k0 < K; k0 += BK) {
        // ---- A tile 128x32 f32 -> packed bf16 pairs -> ds_store_b64 ----
        #pragma unroll
        for (int it = 0; it < 4; it++) {
            int slot = tid + it * 256;            // 0..1023 float4 slots
            int r    = slot >> 3;
            int c4   = (slot & 7) << 2;
            int grow = bm + r;
            float4 v = make_float4(0.f, 0.f, 0.f, 0.f);
            if (grow < Mrows) {
                const float* ap = A + (size_t)grow * lda + k0 + c4;
                v = *(const float4*)ap;
                if (k0 + BK < K) __builtin_prefetch(ap + BK, 0, 1);
            }
            uint2 p;
            p.x = pack_bf16x2(v.x, v.y);
            p.y = pack_bf16x2(v.z, v.w);
            *(uint2*)&sA[r][c4] = p;
        }

        // ---- B tile: overlap next DMA with this step's WMMAs ----
#if __has_builtin(__builtin_amdgcn_tensor_load_to_lds)
        if (wave == 0) {
            if (k0 + BK < K) {
                tdm_issue(k0 + BK, buf ^ 1);           // next tile in flight
                __builtin_amdgcn_s_wait_tensorcnt(1);  // current tile ready
            } else {
                __builtin_amdgcn_s_wait_tensorcnt(0);
            }
        }
#else
        {   // fallback: plain vector copy (Wt already bf16 [N][K])
            int nn = tid >> 2, qq = tid & 3;
            *(uint4*)&sB[buf][nn][qq * 8] =
                *(const uint4*)(Wt + (size_t)(bn + nn) * K + k0 + qq * 8);
        }
#endif
        __syncthreads();

        v16bf afrag[2], bfrag[2];
        #pragma unroll
        for (int t = 0; t < 2; t++) {
            const bf16_t* ap = &sA[wm + t * 16 + mrow][0];
            v8bf alo = *(const v8bf*)(ap + kbA);
            v8bf ahi = *(const v8bf*)(ap + kbA + 16);
            afrag[t] = __builtin_shufflevector(alo, ahi,
                        0,1,2,3,4,5,6,7,8,9,10,11,12,13,14,15);
            const bf16_t* bp = &sB[buf][wn + t * 16 + mrow][0];
            v8bf blo = *(const v8bf*)(bp + kbB);
            v8bf bhi = *(const v8bf*)(bp + kbB + 16);
            bfrag[t] = __builtin_shufflevector(blo, bhi,
                        0,1,2,3,4,5,6,7,8,9,10,11,12,13,14,15);
        }
        #pragma unroll
        for (int tm = 0; tm < 2; tm++)
            #pragma unroll
            for (int tn = 0; tn < 2; tn++)
                acc[tm][tn] = __builtin_amdgcn_wmma_f32_16x16x32_bf16(
                    false, afrag[tm], false, bfrag[tn],
                    (short)0, acc[tm][tn], false, false);
        __syncthreads();
#if __has_builtin(__builtin_amdgcn_tensor_load_to_lds)
        buf ^= 1;
#endif
    }

    // epilogue: D layout VGPR r -> M = r + 8*(lane>=16), N = lane&15
    #pragma unroll
    for (int tm = 0; tm < 2; tm++) {
        #pragma unroll
        for (int tn = 0; tn < 2; tn++) {
            int col     = bn + wn + tn * 16 + (lane & 15);
            int rowHalf = (lane >> 4) * 8;
            #pragma unroll
            for (int r = 0; r < 8; r++) {
                int grow = bm + wm + tm * 16 + rowHalf + r;
                if (grow < Mrows) {
                    float v = acc[tm][tn][r];
                    if (bias) v += bias[col];
                    if (act)  v = gelu_exact(v);
                    C[(size_t)grow * ldc + col] = v;
                }
            }
        }
    }
}

// =====================================================================
// Embedding: h[b,s,:] = LN(x + pos[s] + type[s==0], emb_ln), eps=1e-12
// =====================================================================
__global__ void __launch_bounds__(256)
embed_kernel(const float* __restrict__ feats, const float* __restrict__ cls,
             const float* __restrict__ pos,   const float* __restrict__ typ,
             const float* __restrict__ g,     const float* __restrict__ bt,
             float* __restrict__ h)
{
    int bi = blockIdx.x;
    int b  = bi / Sq;
    int s  = bi - b * Sq;
    int tid = threadIdx.x;
    int c0  = tid * 2;

    float x0, x1;
    if (s == 0) { x0 = cls[c0]; x1 = cls[c0 + 1]; }
    else {
        size_t f = ((size_t)b * Fq + (s - 1)) * Eq;
        x0 = feats[f + c0]; x1 = feats[f + c0 + 1];
    }
    const float* tp = typ + (s == 0 ? Eq : 0);
    x0 += pos[(size_t)s * Eq + c0]     + tp[c0];
    x1 += pos[(size_t)s * Eq + c0 + 1] + tp[c0 + 1];

    float sm = x0 + x1, sq = x0 * x0 + x1 * x1;
    #pragma unroll
    for (int off = 16; off; off >>= 1) {
        sm += __shfl_xor(sm, off, 32);
        sq += __shfl_xor(sq, off, 32);
    }
    __shared__ float rs[8], rq[8];
    int wv = tid >> 5, ln = tid & 31;
    if (ln == 0) { rs[wv] = sm; rq[wv] = sq; }
    __syncthreads();
    float S1 = 0.f, S2 = 0.f;
    #pragma unroll
    for (int w = 0; w < 8; w++) { S1 += rs[w]; S2 += rq[w]; }
    float mean = S1 * (1.0f / Eq);
    float var  = S2 * (1.0f / Eq) - mean * mean;
    float inv  = rsqrtf(var + 1e-12f);
    size_t base = (size_t)bi * Eq;
    h[base + c0]     = (x0 - mean) * inv * g[c0]     + bt[c0];
    h[base + c0 + 1] = (x1 - mean) * inv * g[c0 + 1] + bt[c0 + 1];
}

// =====================================================================
// out = LN(x + a, g, bt, eps); a may be null. One block per row.
// =====================================================================
__global__ void __launch_bounds__(256)
add_ln_kernel(const float* __restrict__ x, const float* __restrict__ a,
              const float* __restrict__ g, const float* __restrict__ bt,
              float* __restrict__ out, float eps)
{
    size_t row = blockIdx.x;
    int tid = threadIdx.x;
    int c0  = tid * 2;
    size_t base = row * Eq;
    float t0 = x[base + c0]     + (a ? a[base + c0]     : 0.f);
    float t1 = x[base + c0 + 1] + (a ? a[base + c0 + 1] : 0.f);
    float sm = t0 + t1, sq = t0 * t0 + t1 * t1;
    #pragma unroll
    for (int off = 16; off; off >>= 1) {
        sm += __shfl_xor(sm, off, 32);
        sq += __shfl_xor(sq, off, 32);
    }
    __shared__ float rs[8], rq[8];
    int wv = tid >> 5, ln = tid & 31;
    if (ln == 0) { rs[wv] = sm; rq[wv] = sq; }
    __syncthreads();
    float S1 = 0.f, S2 = 0.f;
    #pragma unroll
    for (int w = 0; w < 8; w++) { S1 += rs[w]; S2 += rq[w]; }
    float mean = S1 * (1.0f / Eq);
    float var  = S2 * (1.0f / Eq) - mean * mean;
    float inv  = rsqrtf(var + eps);
    out[base + c0]     = (t0 - mean) * inv * g[c0]     + bt[c0];
    out[base + c0 + 1] = (t1 - mean) * inv * g[c0 + 1] + bt[c0 + 1];
}

// =====================================================================
// Banded attention, online softmax. 1 block per (b,i) row, 1 wave per head.
// =====================================================================
__device__ __forceinline__ void attn_step(float q0, float q1, int d0,
    const float* __restrict__ kp, const float* __restrict__ vp,
    float& m, float& l, float& a0, float& a1)
{
    float s = q0 * kp[d0] + q1 * kp[d0 + 1];
    #pragma unroll
    for (int off = 16; off; off >>= 1) s += __shfl_xor(s, off, 32);
    float mn = fmaxf(m, s);
    float c  = __expf(m - mn);
    float p  = __expf(s - mn);
    l  = l * c + p;
    a0 = a0 * c + p * vp[d0];
    a1 = a1 * c + p * vp[d0 + 1];
    m  = mn;
}

__global__ void __launch_bounds__(256)
attn_kernel(const float* __restrict__ q,  const float* __restrict__ k,
            const float* __restrict__ v,  const float* __restrict__ kg,
            const float* __restrict__ vg, const float* __restrict__ qg0,
            float* __restrict__ out)
{
    int bi = blockIdx.x;
    int b  = bi / Sq;
    int i  = bi - b * Sq;
    int hh   = threadIdx.x >> 5;
    int lane = threadIdx.x & 31;
    int d0   = lane * 2;
    size_t rowOff = (size_t)bi * Eq + hh * Dhq;
    size_t base   = (size_t)b * Sq * Eq + hh * Dhq;
    const float SCALE = 0.125f;   // 1/sqrt(64)

    float q0, q1;
    if (i == 0) {
        const float* qp = qg0 + (size_t)b * Eq + hh * Dhq;
        q0 = qp[d0] * SCALE; q1 = qp[d0 + 1] * SCALE;
    } else {
        q0 = q[rowOff + d0] * SCALE; q1 = q[rowOff + d0 + 1] * SCALE;
    }

    float m = -1e30f, l = 0.f, a0 = 0.f, a1 = 0.f;
    if (i == 0) {
        for (int j = 0; j < Sq; j++)
            attn_step(q0, q1, d0, kg + base + (size_t)j * Eq,
                                  vg + base + (size_t)j * Eq, m, l, a0, a1);
    } else {
        int jlo = i - W1Sq; if (jlo < 0)      jlo = 0;
        int jhi = i + W1Sq; if (jhi > Sq - 1) jhi = Sq - 1;
        if (jlo > 0) attn_step(q0, q1, d0, kg + base, vg + base, m, l, a0, a1);
        for (int j = jlo; j <= jhi; j++) {
            const float* kp = (j == 0) ? (kg + base) : (k + base + (size_t)j * Eq);
            const float* vp = (j == 0) ? (vg + base) : (v + base + (size_t)j * Eq);
            attn_step(q0, q1, d0, kp, vp, m, l, a0, a1);
        }
    }
    float inv = 1.f / l;
    out[rowOff + d0]     = a0 * inv;
    out[rowOff + d0 + 1] = a1 * inv;
}

// =====================================================================
// Head output: y[r-1] = t1[r,:] . w2 + b2 for rows 1..287
// =====================================================================
__global__ void __launch_bounds__(256)
headout_kernel(const float* __restrict__ t1, const float* __restrict__ w2,
               const float* __restrict__ b2, float* __restrict__ out)
{
    int r   = blockIdx.x + 1;
    int tid = threadIdx.x;
    size_t base = (size_t)r * Eq;
    float s = t1[base + tid] * w2[tid] + t1[base + tid + 256] * w2[tid + 256];
    #pragma unroll
    for (int off = 16; off; off >>= 1) s += __shfl_xor(s, off, 32);
    __shared__ float red[8];
    if ((tid & 31) == 0) red[tid >> 5] = s;
    __syncthreads();
    if (tid == 0) {
        float tot = 0.f;
        #pragma unroll
        for (int w = 0; w < 8; w++) tot += red[w];
        out[blockIdx.x] = tot + b2[0];
    }
}

// ---------------- host side ----------------
static inline void GEMM(const float* A, int lda, const bf16_t* Wt,
                        const float* bias, float* C, int ldc,
                        int Mr, int N, int K, int act, hipStream_t s)
{
    dim3 grid((Mr + BM - 1) / BM, N / BN);
    gemm_bf16_wmma<<<grid, 256, 0, s>>>(A, lda, Wt, bias, C, ldc, Mr, N, K, act);
}

extern "C" void kernel_launch(void* const* d_in, const int* in_sizes, int n_in,
                              void* d_out, int out_size, void* d_ws, size_t ws_size,
                              hipStream_t stream)
{
    (void)in_sizes; (void)n_in; (void)out_size; (void)ws_size;
    // pytree-sorted leaf order (dict keys alphabetical at every level)
    const float* feats   = (const float*)d_in[0];
    const float* cls     = (const float*)d_in[1];
    const float* embln_b = (const float*)d_in[2];
    const float* embln_g = (const float*)d_in[3];
    const float* h1_b    = (const float*)d_in[4];
    const float* h1_w    = (const float*)d_in[5];
    const float* h2_b    = (const float*)d_in[6];
    const float* h2_w    = (const float*)d_in[7];
    const float* hln_b   = (const float*)d_in[8];
    const float* hln_g   = (const float*)d_in[9];
    const float* pos     = (const float*)d_in[76];
    const float* typ     = (const float*)d_in[77];

    float* ws = (float*)d_ws;
    const size_t ME = (size_t)Mq * Eq;
    float* h    = ws;
    float* qb   = ws + 1 * ME;
    float* kbuf = ws + 2 * ME;
    float* vbuf = ws + 3 * ME;
    float* kgbf = ws + 4 * ME;
    float* vgbf = ws + 5 * ME;
    float* aob  = ws + 6 * ME;
    float* tmp  = ws + 7 * ME;
    float* mid  = qb;                       // reuse q buffer for FFN mid chunk
    float* qg0  = ws + 8 * ME;              // [B,E]
    float* hlnb = qg0 + (size_t)Bq * Eq;    // [S,E]
    float* t1b  = hlnb + (size_t)Sq * Eq;   // [S,E]
    bf16_t* wt_region = (bf16_t*)(t1b + (size_t)Sq * Eq);

    // ---- weight pre-conversion: Wt[N][K] bf16, once per launch ----
    bf16_t* wtcur = wt_region;
    auto conv = [&](const float* W, int K, int N) -> bf16_t* {
        bf16_t* dst = wtcur;
        wtcur += (size_t)K * N;
        dim3 g(K / 32, N / 64);
        wt_convert_kernel<<<g, 256, 0, stream>>>(W, dst, K, N);
        return dst;
    };

    const float *f1b[Lq], *f2b[Lq], *kb[Lq], *kgb[Lq], *l1b[Lq], *l1g[Lq],
                *l2b[Lq], *l2g[Lq], *ob[Lq], *qbs[Lq], *qgb[Lq], *vbs[Lq],
                *vgb[Lq];
    bf16_t *f1t[Lq], *f2t[Lq], *kt[Lq], *kgt[Lq], *ot[Lq], *qt[Lq], *qgt[Lq],
           *vt[Lq], *vgt[Lq];
    for (int l = 0; l < Lq; l++) {
        int o = 10 + 22 * l;
        f1b[l] = (const float*)d_in[o + 0];
        f1t[l] = conv((const float*)d_in[o + 1], Eq, Iq);
        f2b[l] = (const float*)d_in[o + 2];
        f2t[l] = conv((const float*)d_in[o + 3], Iq, Eq);
        kb[l]  = (const float*)d_in[o + 4];
        kt[l]  = conv((const float*)d_in[o + 5], Eq, Eq);
        kgb[l] = (const float*)d_in[o + 6];
        kgt[l] = conv((const float*)d_in[o + 7], Eq, Eq);
        l1b[l] = (const float*)d_in[o + 8];
        l1g[l] = (const float*)d_in[o + 9];
        l2b[l] = (const float*)d_in[o + 10];
        l2g[l] = (const float*)d_in[o + 11];
        ob[l]  = (const float*)d_in[o + 12];
        ot[l]  = conv((const float*)d_in[o + 13], Eq, Eq);
        qbs[l] = (const float*)d_in[o + 14];
        qt[l]  = conv((const float*)d_in[o + 15], Eq, Eq);
        qgb[l] = (const float*)d_in[o + 16];
        qgt[l] = conv((const float*)d_in[o + 17], Eq, Eq);
        vbs[l] = (const float*)d_in[o + 18];
        vt[l]  = conv((const float*)d_in[o + 19], Eq, Eq);
        vgb[l] = (const float*)d_in[o + 20];
        vgt[l] = conv((const float*)d_in[o + 21], Eq, Eq);
    }
    bf16_t* h1t = conv(h1_w, Eq, Eq);

    embed_kernel<<<Mq, 256, 0, stream>>>(feats, cls, pos, typ,
                                         embln_g, embln_b, h);

    for (int l = 0; l < Lq; l++) {
        // projections (M=36864, K=N=512)
        GEMM(h, Eq, qt[l],  qbs[l], qb,   Eq, Mq, Eq, Eq, 0, stream);
        GEMM(h, Eq, kt[l],  kb[l],  kbuf, Eq, Mq, Eq, Eq, 0, stream);
        GEMM(h, Eq, vt[l],  vbs[l], vbuf, Eq, Mq, Eq, Eq, 0, stream);
        GEMM(h, Eq, kgt[l], kgb[l], kgbf, Eq, Mq, Eq, Eq, 0, stream);
        GEMM(h, Eq, vgt[l], vgb[l], vgbf, Eq, Mq, Eq, Eq, 0, stream);
        // qg on cls rows only: A row b at h + b*S*E  ->  lda = S*E, M = B
        GEMM(h, Sq * Eq, qgt[l], qgb[l], qg0, Eq, Bq, Eq, Eq, 0, stream);

        attn_kernel<<<Mq, 256, 0, stream>>>(qb, kbuf, vbuf, kgbf, vgbf,
                                            qg0, aob);

        GEMM(aob, Eq, ot[l], ob[l], tmp, Eq, Mq, Eq, Eq, 0, stream);
        add_ln_kernel<<<Mq, 256, 0, stream>>>(h, tmp, l1g[l], l1b[l], h,
                                              1e-12f);

        // FFN, chunked over rows; GELU fused into first GEMM epilogue
        for (int c = 0; c < Mq / CHq; c++) {
            GEMM(h + (size_t)c * CHq * Eq, Eq, f1t[l], f1b[l],
                 mid, Iq, CHq, Iq, Eq, 1, stream);
            GEMM(mid, Iq, f2t[l], f2b[l],
                 tmp + (size_t)c * CHq * Eq, Eq, CHq, Eq, Iq, 0, stream);
        }
        add_ln_kernel<<<Mq, 256, 0, stream>>>(h, tmp, l2g[l], l2b[l], h,
                                              1e-12f);
    }

    // head: rows 0..287 of batch 0 (= first S rows of h)
    add_ln_kernel<<<Sq, 256, 0, stream>>>(h, nullptr, hln_g, hln_b,
                                          hlnb, 1e-5f);
    GEMM(hlnb, Eq, h1t, h1_b, t1b, Eq, Sq, Eq, Eq, 1, stream);
    headout_kernel<<<Fq, 256, 0, stream>>>(t1b, h2_w, h2_b, (float*)d_out);
}
